// DeepFM_10368051052905
// MI455X (gfx1250) — matrix operand order repaired and checked by reading, not compiled, not run on hardware
//
#include <hip/hip_runtime.h>
#include <hip/hip_bf16.h>

typedef __attribute__((ext_vector_type(16))) _Float16 v16h;
typedef __attribute__((ext_vector_type(8)))  _Float16 v8h;
typedef __attribute__((ext_vector_type(8)))  float    v8f;

#define B_    16384
#define NC_   13
#define F_    26
#define V_    100000
#define D_    16
#define H1_   400
#define H2_   400
#define LDA1  448   // DNN_IN=429 padded to 14*32
#define LDA2  416   // H1=400 padded to 13*32
#define NPW   5     // 16x16 N-tiles per wave (5*16 = 80 cols per workgroup)
#define LDSPAD 8    // halves of per-row LDS padding (bank-conflict destagger)

// ---------------------------------------------------------------------------
// Kernel 1: fused embedding gather + FM (1st & 2nd order) + fp16 pack of dnn_in
// Half-wave (16 lanes) per batch row; lane index == embedding dim d.
// ---------------------------------------------------------------------------
__global__ __launch_bounds__(256) void gather_fm_kernel(
    const float* __restrict__ cont, const int* __restrict__ cat,
    const float* __restrict__ Wc,   const float* __restrict__ bc,
    const float* __restrict__ embF, const float* __restrict__ emb,
    _Float16* __restrict__ dnn, float* __restrict__ fm)
{
  int wave = blockIdx.x * 8 + (threadIdx.x >> 5);
  int lane = threadIdx.x & 31;
  int d = lane & 15;
  int half = lane >> 4;
  int row = wave * 2 + half;                      // grid sized so row < B_

  const int* idxr = cat + row * F_;
  _Float16* drow = dnn + (size_t)row * LDA1;

  float partial = 0.0f;
  if (d < NC_) {
    float c = cont[row * NC_ + d];
    partial += c * Wc[d];
    drow[d] = (_Float16)c;
  }
  // zero K-padding cols 429..447
  drow[429 + d] = (_Float16)0.0f;                 // 429..444
  if (d < 3) drow[445 + d] = (_Float16)0.0f;      // 445..447

  float s = 0.0f, q = 0.0f;
  #pragma unroll 1
  for (int f = 0; f < F_; ++f) {
    int id = idxr[f];
    float e = emb[((size_t)f * V_ + id) * D_ + d];  // 64B coalesced per half-wave
    s += e;
    q += e * e;
    drow[NC_ + f * D_ + d] = (_Float16)e;
    if ((f & 15) == d)                            // distribute emb_first over lanes
      partial += embF[(size_t)f * V_ + id];
  }
  // per-lane contribution: first-order pieces + 0.5*(sum^2 - sumsq) for dim d
  float t = partial + 0.5f * (s * s - q);
  t += __shfl_xor(t, 1, 16);
  t += __shfl_xor(t, 2, 16);
  t += __shfl_xor(t, 4, 16);
  t += __shfl_xor(t, 8, 16);
  if (d == 0) fm[row] = t + bc[0];
}

// ---------------------------------------------------------------------------
// Weight prep: W (K x N, f32 row-major) -> WT (N x ldk, f16, zero K-pad)
// ---------------------------------------------------------------------------
__global__ __launch_bounds__(256) void prep_wt_kernel(
    const float* __restrict__ W, _Float16* __restrict__ WT,
    int K, int N, int ldk)
{
  int i = blockIdx.x * blockDim.x + threadIdx.x;
  if (i >= N * ldk) return;
  int n = i / ldk, k = i % ldk;
  WT[i] = (k < K) ? (_Float16)W[(size_t)k * N + n] : (_Float16)0.0f;
}

// ---------------------------------------------------------------------------
// Panel GEMM: D = relu(A @ BT' + bias).
// Workgroup = 8 waves: stages an 80-col weight panel (BT rows ng*80..+79,
// full K) into LDS via async global->LDS loads, then each wave computes a
// 16x80 strip (5 WMMA tiles, accumulators stay in VGPRs).
// A: M x lda (f16, K zero-padded), BT: N x lda (f16 pre-transposed weights).
// LDS rows padded by LDSPAD halves => lanes hit distinct bank groups.
// ---------------------------------------------------------------------------
__global__ __launch_bounds__(256) void gemm_panel_wmma_kernel(
    const _Float16* __restrict__ A, const _Float16* __restrict__ BT,
    const float* __restrict__ bias,
    _Float16* __restrict__ Dh, float* __restrict__ Df,
    int K, int lda, int ldd, int relu)
{
  extern __shared__ __align__(16) _Float16 smem[];   // [80][lda + LDSPAD]

  const int wid  = threadIdx.x >> 5;
  const int lane = threadIdx.x & 31;
  const int mg = blockIdx.x;          // 0..127  (8 M-tiles each)
  const int ng = blockIdx.y;          // 0..4    (80 cols each)
  const int ldsld = lda + LDSPAD;

  // ---- stage B panel: 80 rows x lda halves (contiguous rows of BT) ----
  const _Float16* panel = BT + (size_t)(ng * (NPW * 16)) * lda;
  const int cpr = lda >> 3;                 // 16B chunks per row
  const int nch = (NPW * 16) * cpr;         // total chunks
#if defined(__gfx1250__)
  for (int i = threadIdx.x; i < nch; i += 256) {
    int r = i / cpr, o = i - r * cpr;
    unsigned dst = (unsigned)(uintptr_t)(smem + (size_t)r * ldsld + o * 8);
    const _Float16* src = panel + (size_t)r * lda + o * 8;
    asm volatile("global_load_async_to_lds_b128 %0, %1, off"
                 :: "v"(dst), "v"(src) : "memory");
  }
  asm volatile("s_wait_asynccnt 0" ::: "memory");
#else
  for (int i = threadIdx.x; i < nch; i += 256) {
    int r = i / cpr, o = i - r * cpr;
    *(v8h*)(smem + (size_t)r * ldsld + o * 8) =
        *(const v8h*)(panel + (size_t)r * lda + o * 8);
  }
#endif
  __syncthreads();

  // ---- compute: one wave -> 16 rows x 80 cols ----
  const int tm   = mg * 8 + wid;            // M tile index (0..1023)
  const int half = lane >> 4;
  const int l15  = lane & 15;
  const _Float16* arow = A + (size_t)(tm * 16 + l15) * lda;

  v8f acc[NPW];
  #pragma unroll
  for (int t = 0; t < NPW; ++t) acc[t] = (v8f){};

  for (int k = 0; k < K; k += 32) {
    // 16-bit A 16x32 layout: elems 0-7 -> K = k+8h.. ; elems 8-15 -> K = k+16+8h..
    v8h a0 = *(const v8h*)(arow + k + 8 * half);
    v8h a1 = *(const v8h*)(arow + k + 16 + 8 * half);
    v16h av = __builtin_shufflevector(a0, a1, 0,1,2,3,4,5,6,7,8,9,10,11,12,13,14,15);
    #pragma unroll
    for (int t = 0; t < NPW; ++t) {
      const _Float16* brow = smem + (size_t)(t * 16 + l15) * ldsld;
      v8h b0 = *(const v8h*)(brow + k + 8 * half);
      v8h b1 = *(const v8h*)(brow + k + 16 + 8 * half);
      v16h bv = __builtin_shufflevector(b0, b1, 0,1,2,3,4,5,6,7,8,9,10,11,12,13,14,15);
      acc[t] = __builtin_amdgcn_wmma_f32_16x16x32_f16(
          false, av, false, bv, (short)0, acc[t], false, false);
    }
  }

  // ---- epilogue: bias + relu, f16 or f32 store ----
  #pragma unroll
  for (int t = 0; t < NPW; ++t) {
    int col = ng * (NPW * 16) + t * 16 + l15;
    float bval = bias ? bias[col] : 0.0f;
    #pragma unroll
    for (int r = 0; r < 8; ++r) {
      float v = acc[t][r] + bval;
      if (relu) v = fmaxf(v, 0.0f);
      size_t off = (size_t)(tm * 16 + 8 * half + r) * ldd + col;
      if (Dh) Dh[off] = (_Float16)v;
      else    Df[off] = v;
    }
  }
}

// ---------------------------------------------------------------------------
// Final fusion: out[b] = fm[b]*W_out[0] + dot(h2[b,:], W_out[1:401]) + b_out
// One wave per row.
// ---------------------------------------------------------------------------
__global__ __launch_bounds__(256) void final_out_kernel(
    const float* __restrict__ h2, const float* __restrict__ fm,
    const float* __restrict__ Wout, const float* __restrict__ bout,
    float* __restrict__ out)
{
  int wave = blockIdx.x * 8 + (threadIdx.x >> 5);
  int lane = threadIdx.x & 31;
  const float* hr = h2 + (size_t)wave * H2_;
  float s = 0.0f;
  for (int j = lane; j < H2_; j += 32)
    s += hr[j] * Wout[1 + j];
  s += __shfl_xor(s, 16, 32);
  s += __shfl_xor(s, 8, 32);
  s += __shfl_xor(s, 4, 32);
  s += __shfl_xor(s, 2, 32);
  s += __shfl_xor(s, 1, 32);
  if (lane == 0) out[wave] = fm[wave] * Wout[0] + s + bout[0];
}

// ---------------------------------------------------------------------------
extern "C" void kernel_launch(void* const* d_in, const int* in_sizes, int n_in,
                              void* d_out, int out_size, void* d_ws, size_t ws_size,
                              hipStream_t stream) {
  const float* cont = (const float*)d_in[0];
  const int*   cat  = (const int*)  d_in[1];
  const float* Wc   = (const float*)d_in[2];
  const float* bc   = (const float*)d_in[3];
  const float* embF = (const float*)d_in[4];
  const float* emb  = (const float*)d_in[5];
  const float* W1   = (const float*)d_in[6];
  const float* b1   = (const float*)d_in[7];
  const float* W2   = (const float*)d_in[8];
  const float* b2   = (const float*)d_in[9];
  const float* Wout = (const float*)d_in[10];
  const float* bout = (const float*)d_in[11];
  float* out = (float*)d_out;

  // Workspace layout (all offsets multiple of 256 B):
  char* ws = (char*)d_ws;
  _Float16* dnn = (_Float16*)(ws + 0);                 // B x 448 f16  = 14,680,064
  _Float16* W1T = (_Float16*)(ws + 14680064);          // 400 x 448 f16 =   358,400
  _Float16* W2T = (_Float16*)(ws + 15038464);          // 400 x 416 f16 =   332,800
  _Float16* h1  = (_Float16*)(ws + 15371264);          // B x 416 f16  = 13,631,488
  float*    h2  = (float*)   (ws + 29002752);          // B x 400 f32  = 26,214,400
  float*    fm  = (float*)   (ws + 55217152);          // B f32        =      65,536

  // Weight transpose+pad to f16
  prep_wt_kernel<<<(H1_ * LDA1 + 255) / 256, 256, 0, stream>>>(W1, W1T, 429, H1_, LDA1);
  prep_wt_kernel<<<(H2_ * LDA2 + 255) / 256, 256, 0, stream>>>(W2, W2T, H1_, H2_, LDA2);

  // Fused gather + FM + fp16 pack (16 rows per block)
  gather_fm_kernel<<<B_ / 16, 256, 0, stream>>>(cont, cat, Wc, bc, embF, emb, dnn, fm);

  // Layer 1: (B x 448) @ (448 x 400) -> relu -> f16.  128 M-groups x 5 N-groups.
  size_t lds1 = (size_t)(NPW * 16) * (LDA1 + LDSPAD) * sizeof(_Float16); // 72,960 B
  gemm_panel_wmma_kernel<<<dim3(B_ / 128, 5), 256, lds1, stream>>>(
      dnn, W1T, b1, h1, nullptr, LDA1, LDA1, LDA2, 1);

  // Layer 2: (B x 416) @ (416 x 400) -> relu -> f32
  size_t lds2 = (size_t)(NPW * 16) * (LDA2 + LDSPAD) * sizeof(_Float16); // 67,840 B
  gemm_panel_wmma_kernel<<<dim3(B_ / 128, 5), 256, lds2, stream>>>(
      h1, W2T, b2, nullptr, h2, LDA2, LDA2, H2_, 1);

  // Fusion + output head
  final_out_kernel<<<B_ / 8, 256, 0, stream>>>(h2, fm, Wout, bout, out);
}
